// FKAConv_40544491274257
// MI455X (gfx1250) — matrix operand order repaired and checked by reading, not compiled, not run on hardware
//
#include <hip/hip_runtime.h>
#include <math.h>

// ---------------- problem constants ----------------
#define BB   4
#define NN   32768
#define MM   8192
#define KK   16
#define DD   128
#define DOUT 128
#define TM   16          // points per tile in the final kernel
#define DCH  32          // d-chunk size (DD/DCH = 4 chunks)

typedef float  v8f  __attribute__((ext_vector_type(8)));
typedef __bf16 v16bf __attribute__((ext_vector_type(16)));

// ---------------- workspace layout (bytes) ----------------
// [0,2048): scalars + stats
//   f[0] = rsum (sum of per-point max radius), f[1] = inv_nr
//   stats1: 128 floats at float-offset 64   ((b*16+o)*2 + {sum,sumsq})
//   stats2: 128 floats at float-offset 192
#define WS_SCAL   0u
#define WS_PTS    2048u                                   // B*M*K*3 f32 = 6291456
#define WS_DW     (WS_PTS + (size_t)BB*MM*KK*3*4)         // B*M*K f32  = 2097152
#define WS_CVB    (WS_DW  + (size_t)BB*MM*KK*4)           // 128*2048 bf16 = 524288
#define WS_XT     (WS_CVB + (size_t)DOUT*DD*KK*2)         // B*N*D bf16 = 33554432

__device__ __forceinline__ unsigned short f2bf(float f) {
    union { float f; unsigned int u; } v; v.f = f;
    unsigned int u = v.u;
    unsigned int r = u + 0x7FFFu + ((u >> 16) & 1u);
    return (unsigned short)(r >> 16);
}
__device__ __forceinline__ float bf2f(unsigned short h) {
    union { unsigned int u; float f; } v; v.u = ((unsigned int)h) << 16;
    return v.f;
}

// ---------------- K0: zero the accumulator region ----------------
__global__ void k_init(float* scal) {
    scal[threadIdx.x] = 0.0f;     // 512 floats = 2048 bytes
}

// ---------------- K_cv: cv_w f32 -> bf16, layout [o][d*16+j] ----------------
__global__ void k_cv(const float* __restrict__ cvw, unsigned short* __restrict__ cvb) {
    int i = blockIdx.x * 256 + threadIdx.x;      // 128*128*16 = 262144
    cvb[i] = f2bf(cvw[i]);                       // cv_w is (o,d,j) row-major == [o][dj]
}

// ---------------- K_xt: x (B,D,N) f32 -> xT (B,N,D) bf16 ----------------
__global__ void k_xt(const float* __restrict__ x, unsigned int* __restrict__ xtu) {
    int t = blockIdx.x * 256 + threadIdx.x;      // B*N = 131072 threads
    int b = t / NN, n = t % NN;
    size_t obase = (size_t)(b * NN + n) * (DD / 2);
#pragma unroll 4
    for (int d2 = 0; d2 < DD / 2; ++d2) {
        float lo = x[(size_t)(b * DD + 2 * d2) * NN + n];
        float hi = x[(size_t)(b * DD + 2 * d2 + 1) * NN + n];
        xtu[obase + d2] = (unsigned int)f2bf(lo) | ((unsigned int)f2bf(hi) << 16);
    }
}

// ---------------- K1: pts, dist, dw, radius reduction ----------------
__global__ __launch_bounds__(256) void k1_prep(
    const float* __restrict__ xyz, const float* __restrict__ xyzc,
    const int* __restrict__ idx, const float* __restrict__ alphap,
    const float* __restrict__ betap, float* __restrict__ pts,
    float* __restrict__ dw, float* __restrict__ scal) {
    __shared__ float red;
    if (threadIdx.x == 0) red = 0.0f;
    __syncthreads();

    int p = blockIdx.x * 256 + threadIdx.x;      // point id in [0, B*M)
    int b = p / MM;
    float alpha = alphap[0], beta = betap[0];
    float c0 = xyzc[p * 3 + 0], c1 = xyzc[p * 3 + 1], c2 = xyzc[p * 3 + 2];

    float dwv[KK];
    float dmax = 0.0f, ssum = 0.0f;
#pragma unroll
    for (int k = 0; k < KK; ++k) {
        int n = idx[p * KK + k];
        size_t gx = (size_t)(b * NN + n) * 3;
        float px = xyz[gx + 0] - c0;
        float py = xyz[gx + 1] - c1;
        float pz = xyz[gx + 2] - c2;
        size_t po = (size_t)(p * KK + k) * 3;
        pts[po + 0] = px; pts[po + 1] = py; pts[po + 2] = pz;
        float d = sqrtf(px * px + py * py + pz * pz);
        dmax = fmaxf(dmax, d);
        float z = beta - alpha * d;
        float sg = 1.0f / (1.0f + expf(-z));
        dwv[k] = sg;
        ssum += sg;
    }
    float s = ssum + (ssum == 0.0f ? 1.0f : 0.0f) + 1e-6f;
    float scale = (float)KK / s;
#pragma unroll
    for (int k = 0; k < KK; ++k) dw[p * KK + k] = dwv[k] * scale;

    atomicAdd(&red, dmax);
    __syncthreads();
    if (threadIdx.x == 0) atomicAdd(&scal[0], red);
}

// ---------------- K2: finalize inverse norm radius ----------------
__global__ void k2_nr(const float* __restrict__ nrp, float* __restrict__ scal) {
    float mr = scal[0] * (1.0f / (float)(BB * MM));
    float nr = nrp[0] * 0.9f + mr * 0.1f;
    scal[1] = 1.0f / nr;
}

// ---------------- K3: inorm-1 statistics ----------------
__global__ __launch_bounds__(256) void k3_stats1(
    const float* __restrict__ pts, const float* __restrict__ scal,
    const float* __restrict__ w1, float* __restrict__ stats1) {
    __shared__ float s_pts[TM][KK][3];
    __shared__ float s_acc[KK][2];
    int tid = threadIdx.x;
    int b = blockIdx.x / (MM / TM);
    int m0 = (blockIdx.x % (MM / TM)) * TM;
    float inv_nr = scal[1];

    for (int i = tid; i < TM * KK * 3; i += 256)
        (&s_pts[0][0][0])[i] = pts[(size_t)(b * MM + m0) * KK * 3 + i] * inv_nr;
    if (tid < 32) s_acc[tid >> 1][tid & 1] = 0.0f;
    __syncthreads();

    int m = tid >> 4, o = tid & 15;
    float a = w1[o * 3 + 0], bw = w1[o * 3 + 1], cw = w1[o * 3 + 2];
    float su = 0.0f, sq = 0.0f;
#pragma unroll
    for (int k = 0; k < KK; ++k) {
        float y = s_pts[m][k][0] * a + s_pts[m][k][1] * bw + s_pts[m][k][2] * cw;
        su += y; sq += y * y;
    }
    atomicAdd(&s_acc[o][0], su);
    atomicAdd(&s_acc[o][1], sq);
    __syncthreads();
    if (tid < 32)
        atomicAdd(&stats1[(b * 16 + (tid >> 1)) * 2 + (tid & 1)], s_acc[tid >> 1][tid & 1]);
}

// ---------------- K4: inorm-2 statistics (recompute mat1 -> lin2) ----------------
__global__ __launch_bounds__(256) void k4_stats2(
    const float* __restrict__ pts, const float* __restrict__ dw,
    const float* __restrict__ scal, const float* __restrict__ w1,
    const float* __restrict__ g1, const float* __restrict__ b1,
    const float* __restrict__ w2, const float* __restrict__ stats1,
    float* __restrict__ stats2) {
    __shared__ float s_pts[TM][KK][3];
    __shared__ float s_dw[TM][KK];
    __shared__ float s_m1[TM][KK][KK];
    __shared__ float s_mp[TM][KK];
    __shared__ float s_acc[KK][2];
    int tid = threadIdx.x;
    int b = blockIdx.x / (MM / TM);
    int m0 = (blockIdx.x % (MM / TM)) * TM;
    float inv_nr = scal[1];
    const float inv_cnt = 1.0f / (float)(MM * KK);

    for (int i = tid; i < TM * KK * 3; i += 256)
        (&s_pts[0][0][0])[i] = pts[(size_t)(b * MM + m0) * KK * 3 + i] * inv_nr;
    (&s_dw[0][0])[tid] = dw[(size_t)(b * MM + m0) * KK + tid];
    if (tid < 32) s_acc[tid >> 1][tid & 1] = 0.0f;
    __syncthreads();

    int m = tid >> 4, o = tid & 15;
    // mat1 = relu(inorm1(pts @ w1))
    {
        float mu = stats1[(b * 16 + o) * 2 + 0] * inv_cnt;
        float ex2 = stats1[(b * 16 + o) * 2 + 1] * inv_cnt;
        float var = fmaxf(ex2 - mu * mu, 0.0f);
        float sc = rsqrtf(var + 1e-5f) * g1[o];
        float bb = b1[o];
        float a = w1[o * 3 + 0], bw = w1[o * 3 + 1], cw = w1[o * 3 + 2];
        float mp = 0.0f;
#pragma unroll
        for (int k = 0; k < KK; ++k) {
            float y = s_pts[m][k][0] * a + s_pts[m][k][1] * bw + s_pts[m][k][2] * cw;
            float v = fmaxf((y - mu) * sc + bb, 0.0f);
            s_m1[m][o][k] = v;
            mp = fmaxf(mp, v * s_dw[m][k]);
        }
        s_mp[m][o] = mp;
    }
    __syncthreads();
    // lin2 = [mat1; mp1] @ w2^T : accumulate inorm2 stats
    {
        float addc = 0.0f;
#pragma unroll
        for (int c = 0; c < KK; ++c) addc += s_mp[m][c] * w2[o * 32 + 16 + c];
        float su = 0.0f, sq = 0.0f;
#pragma unroll
        for (int k = 0; k < KK; ++k) {
            float acc = addc;
#pragma unroll
            for (int c = 0; c < KK; ++c) acc += s_m1[m][c][k] * w2[o * 32 + c];
            su += acc; sq += acc * acc;
        }
        atomicAdd(&s_acc[o][0], su);
        atomicAdd(&s_acc[o][1], sq);
    }
    __syncthreads();
    if (tid < 32)
        atomicAdd(&stats2[(b * 16 + (tid >> 1)) * 2 + (tid & 1)], s_acc[tid >> 1][tid & 1]);
}

// ---------------- K5: fused MLP + WMMA contraction ----------------
__global__ __launch_bounds__(256) void k5_final(
    const int* __restrict__ idx, const float* __restrict__ pts,
    const float* __restrict__ dw, const float* __restrict__ scal,
    const float* __restrict__ stats1, const float* __restrict__ stats2,
    const float* __restrict__ w1, const float* __restrict__ g1,
    const float* __restrict__ b1, const float* __restrict__ w2,
    const float* __restrict__ g2, const float* __restrict__ b2,
    const float* __restrict__ w3, const unsigned short* __restrict__ xT,
    const unsigned short* __restrict__ cvb, float* __restrict__ out) {
    __shared__ float s_pts[TM][KK][3];                    //  3 KB
    __shared__ float s_dw[TM][KK];                        //  1 KB
    __shared__ float s_mp[TM][KK];                        //  1 KB
    __shared__ float s_bufA[TM][KK][KK];                  // 16 KB : mat1 then mat3
    __shared__ alignas(16) float s_bufB[TM][KK][KK];      // 16 KB : mat2, reused as xg
    __shared__ unsigned int s_tmp[256][TM];               // 16 KB : packed bf16 pairs

    int tid = threadIdx.x;
    int b = blockIdx.x / (MM / TM);
    int m0 = (blockIdx.x % (MM / TM)) * TM;
    float inv_nr = scal[1];
    const float inv_cnt = 1.0f / (float)(MM * KK);

    for (int i = tid; i < TM * KK * 3; i += 256)
        (&s_pts[0][0][0])[i] = pts[(size_t)(b * MM + m0) * KK * 3 + i] * inv_nr;
    (&s_dw[0][0])[tid] = dw[(size_t)(b * MM + m0) * KK + tid];
    __syncthreads();

    int m = tid >> 4, o = tid & 15;
    // ---- stage 1: mat1 + mp1 ----
    {
        float mu = stats1[(b * 16 + o) * 2 + 0] * inv_cnt;
        float ex2 = stats1[(b * 16 + o) * 2 + 1] * inv_cnt;
        float var = fmaxf(ex2 - mu * mu, 0.0f);
        float sc = rsqrtf(var + 1e-5f) * g1[o];
        float bb = b1[o];
        float a = w1[o * 3 + 0], bw = w1[o * 3 + 1], cw = w1[o * 3 + 2];
        float mp = 0.0f;
#pragma unroll
        for (int k = 0; k < KK; ++k) {
            float y = s_pts[m][k][0] * a + s_pts[m][k][1] * bw + s_pts[m][k][2] * cw;
            float v = fmaxf((y - mu) * sc + bb, 0.0f);
            s_bufA[m][o][k] = v;
            mp = fmaxf(mp, v * s_dw[m][k]);
        }
        s_mp[m][o] = mp;
    }
    __syncthreads();
    // ---- stage 2: lin2 -> mat2 + mp2 ----
    float lin[KK];
    {
        float addc = 0.0f;
#pragma unroll
        for (int c = 0; c < KK; ++c) addc += s_mp[m][c] * w2[o * 32 + 16 + c];
#pragma unroll
        for (int k = 0; k < KK; ++k) {
            float acc = addc;
#pragma unroll
            for (int c = 0; c < KK; ++c) acc += s_bufA[m][c][k] * w2[o * 32 + c];
            lin[k] = acc;
        }
    }
    __syncthreads();   // mp1 fully consumed before overwrite
    {
        float mu = stats2[(b * 16 + o) * 2 + 0] * inv_cnt;
        float ex2 = stats2[(b * 16 + o) * 2 + 1] * inv_cnt;
        float var = fmaxf(ex2 - mu * mu, 0.0f);
        float sc = rsqrtf(var + 1e-5f) * g2[o];
        float bb = b2[o];
        float mp = 0.0f;
#pragma unroll
        for (int k = 0; k < KK; ++k) {
            float v = fmaxf((lin[k] - mu) * sc + bb, 0.0f);
            s_bufB[m][o][k] = v;
            mp = fmaxf(mp, v * s_dw[m][k]);
        }
        s_mp[m][o] = mp;
    }
    __syncthreads();
    // ---- stage 3: mat3 = relu([mat2; mp2] @ w3^T) * dw  (into bufA) ----
    {
        int j = o;
        float addc = 0.0f;
#pragma unroll
        for (int c = 0; c < KK; ++c) addc += s_mp[m][c] * w3[j * 32 + 16 + c];
#pragma unroll
        for (int k = 0; k < KK; ++k) {
            float acc = addc;
#pragma unroll
            for (int c = 0; c < KK; ++c) acc += s_bufB[m][c][k] * w3[j * 32 + c];
            s_bufA[m][j][k] = fmaxf(acc, 0.0f) * s_dw[m][k];
        }
    }
    __syncthreads();

    // ---- stage 4: d-chunked gather + tmp + WMMA ----
    int wave = tid >> 5, lane = tid & 31;
    int lane15 = lane & 15;
    int base8 = (lane & 16) ? 8 : 0;
    int o_row = (wave << 4) + lane15;                 // A-matrix row this lane holds
    v8f accC = {};

    int kkk = tid & 15;                               // gather role: (m, kkk)
    int n_idx = idx[(size_t)(b * MM + m0 + m) * KK + kkk];
    const uint4* xrow = (const uint4*)(xT + (size_t)(b * NN + n_idx) * DD);
    uint4* s_xg4 = (uint4*)&s_bufB[0][0][0];
    const unsigned short* s_xg = (const unsigned short*)&s_bufB[0][0][0];
    const uint4* cv4 = (const uint4*)cvb;

    for (int ch = 0; ch < DD / DCH; ++ch) {
        // gather this 32-d chunk of the 256 neighbor rows (64 B contiguous each)
#pragma unroll
        for (int q = 0; q < 4; ++q)
            s_xg4[(m * KK + kkk) * 4 + q] = xrow[ch * 4 + q];
        __syncthreads();

        // tmp[dj][m] for dj in this chunk, packed as bf16 pairs (B-matrix rows 2p,2p+1)
        int mt = tid & 15;
#pragma unroll
        for (int i = 0; i < 16; ++i) {
            int pl = i * 16 + (tid >> 4);             // pair index in [0,256)
            int dloc = pl >> 3;                       // local d in [0,32)
            int j0 = (pl & 7) * 2;
            float a0 = 0.0f, a1 = 0.0f;
#pragma unroll
            for (int k = 0; k < KK; ++k) {
                float xv = bf2f(s_xg[(mt * KK + k) * DCH + dloc]);
                a0 += xv * s_bufA[mt][j0][k];
                a1 += xv * s_bufA[mt][j0 + 1][k];
            }
            s_tmp[pl][mt] = (unsigned int)f2bf(a0) | ((unsigned int)f2bf(a1) << 16);
        }
        __syncthreads();

        // 16 WMMA k-steps over this chunk; wave w owns output rows [16w,16w+16)
        for (int kl = 0; kl < 16; ++kl) {
            union { v16bf v; uint4 q[2]; } af;
            size_t e = (size_t)o_row * 2048 + (size_t)(ch * 16 + kl) * 32 + base8 * 2;
            af.q[0] = cv4[e >> 3];
            af.q[1] = cv4[(e >> 3) + 1];
            union { v16bf v; unsigned int u[8]; } bf;
#pragma unroll
            for (int p = 0; p < 8; ++p)
                bf.u[p] = s_tmp[kl * 16 + base8 + p][lane15];
            accC = __builtin_amdgcn_wmma_f32_16x16x32_bf16(
                false, af.v, false, bf.v, (short)0, accC, false, false);
        }
        __syncthreads();
    }

    // ---- epilogue: C layout -> out (B, DO, M) ----
#pragma unroll
    for (int r = 0; r < 8; ++r) {
        int oo = (wave << 4) + r + ((lane & 16) ? 8 : 0);
        out[((size_t)b * DOUT + oo) * MM + m0 + lane15] = accC[r];
    }
}

// ---------------- host launcher ----------------
extern "C" void kernel_launch(void* const* d_in, const int* in_sizes, int n_in,
                              void* d_out, int out_size, void* d_ws, size_t ws_size,
                              hipStream_t stream) {
    const float* xyz   = (const float*)d_in[0];
    const float* x     = (const float*)d_in[1];
    const float* xyzc  = (const float*)d_in[2];
    const int*   idx   = (const int*)d_in[3];
    const float* alpha = (const float*)d_in[4];
    const float* beta  = (const float*)d_in[5];
    const float* nrp   = (const float*)d_in[6];
    const float* w1    = (const float*)d_in[7];
    const float* g1    = (const float*)d_in[8];
    const float* b1    = (const float*)d_in[9];
    const float* w2    = (const float*)d_in[10];
    const float* g2    = (const float*)d_in[11];
    const float* b2    = (const float*)d_in[12];
    const float* w3    = (const float*)d_in[13];
    const float* cvw   = (const float*)d_in[14];
    float* out = (float*)d_out;

    char* ws = (char*)d_ws;
    float*          scal   = (float*)(ws + WS_SCAL);
    float*          stats1 = scal + 64;
    float*          stats2 = scal + 192;
    float*          pts    = (float*)(ws + WS_PTS);
    float*          dw     = (float*)(ws + WS_DW);
    unsigned short* cvb    = (unsigned short*)(ws + WS_CVB);
    unsigned short* xT     = (unsigned short*)(ws + WS_XT);

    k_init<<<1, 512, 0, stream>>>(scal);
    k_cv<<<(DOUT * DD * KK) / 256, 256, 0, stream>>>(cvw, cvb);
    k_xt<<<(BB * NN) / 256, 256, 0, stream>>>(x, (unsigned int*)xT);
    k1_prep<<<(BB * MM) / 256, 256, 0, stream>>>(xyz, xyzc, idx, alpha, beta, pts, dw, scal);
    k2_nr<<<1, 1, 0, stream>>>(nrp, scal);
    k3_stats1<<<BB * (MM / TM), 256, 0, stream>>>(pts, scal, w1, stats1);
    k4_stats2<<<BB * (MM / TM), 256, 0, stream>>>(pts, dw, scal, w1, g1, b1, w2, stats1, stats2);
    k5_final<<<BB * (MM / TM), 256, 0, stream>>>(idx, pts, dw, scal, stats1, stats2,
                                                 w1, g1, b1, w2, g2, b2, w3, xT, cvb, out);
}
